// W8A8OFP32LinearWithSFactorCublas_32203664785513
// MI455X (gfx1250) — 8-bit matrix operand order repaired and checked by reading, compiled, not run on hardware
//
#include <hip/hip_runtime.h>
#include <stdint.h>

typedef __attribute__((ext_vector_type(8))) int          v8i;
typedef __attribute__((ext_vector_type(4))) unsigned int v4u;

#define M_TOT 8192
#define N_TOT 4096
#define K_TOT 4096
#define TM 128
#define TN 128
#define TK 64
#define LDSP 80
#define ABUF_SZ (TM * LDSP)

__global__ __launch_bounds__(256)
void quant_kernel(const float* __restrict__ x, signed char* __restrict__ q,
                  const float* __restrict__ inscale_p)
{
    const float inv = 1.0f / inscale_p[0];
    const size_t base = ((size_t)blockIdx.x * 256u + threadIdx.x) * 16u;
    const float4* xp = (const float4*)(x + base);
    v4u outw;
#pragma unroll
    for (int g = 0; g < 4; ++g) {
        float4 f = xp[g];
        int a = (int)__builtin_rintf(f.x * inv);
        int b = (int)__builtin_rintf(f.y * inv);
        int c = (int)__builtin_rintf(f.z * inv);
        int d = (int)__builtin_rintf(f.w * inv);
        a = a < -128 ? -128 : (a > 127 ? 127 : a);
        b = b < -128 ? -128 : (b > 127 ? 127 : b);
        c = c < -128 ? -128 : (c > 127 ? 127 : c);
        d = d < -128 ? -128 : (d > 127 ? 127 : d);
        outw[g] = (unsigned)(a & 255) | ((unsigned)(b & 255) << 8) |
                  ((unsigned)(c & 255) << 16) | ((unsigned)(d & 255) << 24);
    }
    *(v4u*)(q + base) = outw;
}

__global__ __launch_bounds__(256)
void w8a8_gemm_kernel(const signed char* __restrict__ qa,
                      const signed char* __restrict__ wt,
                      float* __restrict__ out,
                      const float* __restrict__ alpha_p)
{
    __shared__ __align__(16) signed char As[2][TM][LDSP];
    __shared__ __align__(16) signed char Bs[2][TN][LDSP];

    const unsigned tid    = threadIdx.x;
    const unsigned lane   = tid & 31u;
    const unsigned wid    = tid >> 5;
    const unsigned lane16 = lane & 15u;
    const unsigned half   = lane >> 4;
    const unsigned m_base = (wid >> 1) * 32u;
    const unsigned n_base = (wid & 1u) * 64u;

    const unsigned m0 = blockIdx.y * TM;
    const unsigned n0 = blockIdx.x * TN;

    const unsigned as_base = (unsigned)(uintptr_t)&As[0][0][0];
    const unsigned bs_base = (unsigned)(uintptr_t)&Bs[0][0][0];

    const unsigned row0 = tid >> 2;
    const unsigned col0 = (tid & 3u) * 16u;
    const unsigned row1 = row0 + 64u;

    const unsigned lA0 = as_base + row0 * LDSP + col0;
    const unsigned lA1 = as_base + row1 * LDSP + col0;
    const unsigned lB0 = bs_base + row0 * LDSP + col0;
    const unsigned lB1 = bs_base + row1 * LDSP + col0;
    unsigned gA0 = (m0 + row0) * K_TOT + col0;
    unsigned gA1 = (m0 + row1) * K_TOT + col0;
    unsigned gB0 = (n0 + row0) * K_TOT + col0;
    unsigned gB1 = (n0 + row1) * K_TOT + col0;

    auto prefetch = [&](unsigned boff) {
        asm volatile("global_load_async_to_lds_b128 %0, %1, %2"
                     :: "v"(lA0 + boff), "v"(gA0), "s"(qa) : "memory");
        asm volatile("global_load_async_to_lds_b128 %0, %1, %2"
                     :: "v"(lA1 + boff), "v"(gA1), "s"(qa) : "memory");
        asm volatile("global_load_async_to_lds_b128 %0, %1, %2"
                     :: "v"(lB0 + boff), "v"(gB0), "s"(wt) : "memory");
        asm volatile("global_load_async_to_lds_b128 %0, %1, %2"
                     :: "v"(lB1 + boff), "v"(gB1), "s"(wt) : "memory");
        gA0 += TK; gA1 += TK; gB0 += TK; gB1 += TK;
    };

    v8i acc[2][4];
#pragma unroll
    for (int i = 0; i < 2; ++i)
#pragma unroll
        for (int j = 0; j < 4; ++j)
            acc[i][j] = (v8i)0;

    unsigned a_rel[2], b_rel[4];
#pragma unroll
    for (int i = 0; i < 2; ++i)
        a_rel[i] = (m_base + i * 16u + lane16) * LDSP + half * 8u;
#pragma unroll
    for (int j = 0; j < 4; ++j)
        b_rel[j] = (n_base + j * 16u + lane16) * LDSP + half * 8u;

    const signed char* a_sp = &As[0][0][0];
    const signed char* b_sp = &Bs[0][0][0];

    prefetch(0);

    unsigned cur = 0;
    const unsigned NKT = K_TOT / TK;
#pragma unroll 1
    for (unsigned kt = 0; kt < NKT; ++kt) {
        asm volatile("s_wait_asynccnt 0x0" ::: "memory");
        __syncthreads();
        if (kt + 1u < NKT) prefetch(cur ^ ABUF_SZ);

        union { v8i v; unsigned long long d[4]; } a[2];
#pragma unroll
        for (int i = 0; i < 2; ++i) {
            const signed char* rp = a_sp + (a_rel[i] + cur);
#pragma unroll
            for (int g = 0; g < 4; ++g)
                a[i].d[g] = *(const unsigned long long*)(rp + g * 16);
        }
        union { v8i v; unsigned long long d[4]; } b[4];
#pragma unroll
        for (int j = 0; j < 4; ++j) {
            const signed char* rp = b_sp + (b_rel[j] + cur);
#pragma unroll
            for (int p = 0; p < 4; ++p)
                b[j].d[p] = *(const unsigned long long*)(rp + p * 16);
        }

#pragma unroll
        for (int i = 0; i < 2; ++i)
#pragma unroll
            for (int j = 0; j < 4; ++j)
                acc[i][j] = __builtin_amdgcn_wmma_i32_16x16x64_iu8(
                     true, a[i].v,  true, b[j].v,
                    acc[i][j],  false,  false);

        cur ^= ABUF_SZ;
    }

    const float alpha = alpha_p[0];
#pragma unroll
    for (int i = 0; i < 2; ++i) {
#pragma unroll
        for (int j = 0; j < 4; ++j) {
            const unsigned gn = n0 + n_base + j * 16u + lane16;
#pragma unroll
            for (int r = 0; r < 8; ++r) {
                const unsigned gm = m0 + m_base + i * 16u + half * 8u + (unsigned)r;
                out[(size_t)gm * N_TOT + gn] = alpha * (float)acc[i][j][r];
            }
        }
    }
}

extern "C" void kernel_launch(void* const* d_in, const int* in_sizes, int n_in,
                              void* d_out, int out_size, void* d_ws, size_t ws_size,
                              hipStream_t stream) {
    const float*       x       = (const float*)d_in[0];
    const signed char* weight  = (const signed char*)d_in[1];
    const float*       alpha   = (const float*)d_in[2];
    const float*       inscale = (const float*)d_in[3];
    float*             out     = (float*)d_out;
    signed char*       q       = (signed char*)d_ws;

    const unsigned qblocks = (unsigned)(((size_t)M_TOT * K_TOT) / (256u * 16u));
    quant_kernel<<<qblocks, 256, 0, stream>>>(x, q, inscale);

    dim3 grid(N_TOT / TN, M_TOT / TM);
    w8a8_gemm_kernel<<<grid, 256, 0, stream>>>(q, weight, out, alpha);
}
